// CtxAttacker_69801808495249
// MI455X (gfx1250) — compile-verified, matrix operand order checked by reading, not run on hardware
//
#include <hip/hip_runtime.h>
#include <hip/hip_bf16.h>

typedef __attribute__((ext_vector_type(16))) _Float16 v16h;
typedef __attribute__((ext_vector_type(8)))  float    v8f;
typedef __attribute__((ext_vector_type(4)))  unsigned int v4u;
typedef __attribute__((ext_vector_type(8)))  int      v8i;
typedef __attribute__((ext_vector_type(4)))  int      v4i;

#define B_  32
#define L_  128
#define H_  512
#define GN_ 2048   // 4*H
#define D2_ 1024   // 2*H
#define V_  32000
#define E_  128

__device__ __forceinline__ v8f wmma_f16(v16h a, v16h b, v8f c) {
  return __builtin_amdgcn_wmma_f32_16x16x32_f16(false, a, false, b, (short)0, c, false, false);
}

// ---------------- utility kernels ----------------

__global__ void init_accum_kernel(float* accum) {
  accum[0] = 0.f; accum[1] = 0.f;
}

__global__ void finalize_kernel(float* out, const float* accum) {
  out[0] = accum[0] / accum[1];
}

__global__ void f32_to_f16_kernel(const float* __restrict__ in, _Float16* __restrict__ outp, int n) {
  int i = blockIdx.x * 256 + threadIdx.x;
  if (i < n) outp[i] = (_Float16)in[i];
}

__global__ void embed_f16_kernel(const int* __restrict__ word, const float* __restrict__ emb,
                                 _Float16* __restrict__ x, int rows, int e) {
  int i = blockIdx.x * 256 + threadIdx.x;
  if (i < rows * e) {
    int r = i / e, c = i - r * e;
    x[i] = (_Float16)emb[(size_t)word[r] * e + c];
  }
}

// ---------------- generic f16 WMMA GEMM: C[M,N] = A[M,K] * W[N,K]^T + bias ----------------
// The 16xK A-tile (shared by all 8 waves of the block) is staged into LDS once via the
// Tensor Data Mover (tensor_load_to_lds + s_wait_tensorcnt); each wave then computes one
// 16x16 output tile with v_wmma_f32_16x16x32_f16, streaming W from L2.
__global__ __launch_bounds__(256) void gemm_f16_kernel(
    const _Float16* __restrict__ A, const _Float16* __restrict__ W,
    const float* __restrict__ bias, float* __restrict__ Cf,
    _Float16* __restrict__ Ch, int M, int N, int K, int relu)
{
  __shared__ __align__(128) _Float16 atile[16 * 1024];   // K <= 1024

  const int lane = threadIdx.x & 31;
  const int wave = threadIdx.x >> 5;
  const int m0 = blockIdx.x * 16;
  const int n0 = blockIdx.y * 128 + wave * 16;

#if __has_builtin(__builtin_amdgcn_tensor_load_to_lds)
  if (wave == 0) {
    // Build Tensor DMA descriptor (D#): 2D tile, 16 rows x K f16 elements, row stride K.
    const uint64_t ga   = (uint64_t)(uintptr_t)(A + (size_t)m0 * K);
    const uint32_t galo = (uint32_t)ga;
    const uint32_t gahi = (uint32_t)(ga >> 32);
    const uint32_t lds  = (uint32_t)(uintptr_t)(&atile[0]);
    v4u g0;
    g0[0] = 1u;                                   // count=1, user descriptor
    g0[1] = lds;                                  // lds_addr
    g0[2] = galo;                                 // global_addr[31:0]
    g0[3] = (gahi & 0x01FFFFFFu) | (2u << 30);    // global_addr[56:32], type=2 (image)
    const uint32_t dim0 = (uint32_t)K;            // tensor_dim0 (elements)
    const uint32_t dim1 = 16u;                    // tensor_dim1 (rows)
    const uint64_t str0 = (uint64_t)K;            // tensor_dim0_stride
    v8i g1;
    g1[0] = (int)(1u << 16);                                        // data_size=2B
    g1[1] = (int)((dim0 & 0xFFFFu) << 16);                          // tensor_dim0[15:0]
    g1[2] = (int)((dim0 >> 16) | ((dim1 & 0xFFFFu) << 16));         // dim0 hi | dim1 lo
    g1[3] = (int)((dim1 >> 16) | (((uint32_t)K & 0xFFFFu) << 16));  // dim1 hi | tile_dim0
    g1[4] = (int)16u;                                               // tile_dim1=16, tile_dim2=0
    g1[5] = (int)(uint32_t)(str0 & 0xFFFFFFFFu);                    // stride0[31:0]
    g1[6] = (int)(uint32_t)((str0 >> 32) & 0xFFFFu);                // stride0[47:32]
    g1[7] = 0;
    v4i z4 = {0, 0, 0, 0};
    v8i z8 = {0, 0, 0, 0, 0, 0, 0, 0};
    __builtin_amdgcn_tensor_load_to_lds(g0, g1, z4, z4, z8, 0);
  }
  __builtin_amdgcn_s_wait_tensorcnt(0);
  __syncthreads();
#else
  for (int i = threadIdx.x; i < 16 * K; i += 256)
    atile[i] = A[(size_t)m0 * K + i];
  __syncthreads();
#endif

  const _Float16* arow = atile + (size_t)(lane & 15) * K + ((lane >> 4) * 16);
  const _Float16* wrow = W + (size_t)(n0 + (lane & 15)) * K + ((lane >> 4) * 16);
  v8f acc = {};
  for (int k0 = 0; k0 < K; k0 += 32) {
    v16h a = *(const v16h*)(arow + k0);
    v16h b = *(const v16h*)(wrow + k0);
    acc = wmma_f16(a, b, acc);
  }
  const int col = n0 + (lane & 15);
  const float bb = bias ? bias[col] : 0.f;
#pragma unroll
  for (int v = 0; v < 8; ++v) {
    int row = m0 + v + 8 * (lane >> 4);
    float val = acc[v] + bb;
    if (relu) val = val > 0.f ? val : 0.f;
    if (Cf) Cf[(size_t)row * N + col] = val;
    if (Ch) Ch[(size_t)row * N + col] = (_Float16)val;
  }
}

// ---------------- persistent recurrent LSTM kernel: one workgroup per direction ----------------
// gx  : [2][B][L][GN] f32 (input projection + bias, per direction)
// whh : [2][GN][H]    f16
// out : [B][L][2H]    f16  (direction writes its H-column half)
// Dynamic LDS: gates f16 [B][GN] (128KB) + h f16 [B][H] (32KB) = 160KB.
__global__ __launch_bounds__(1024) void lstm_rec_kernel(
    const float* __restrict__ gx2, const _Float16* __restrict__ whh2,
    const float* __restrict__ mask, _Float16* __restrict__ outp)
{
  extern __shared__ char smem[];
  _Float16* gatesL = (_Float16*)smem;                                 // [32][2048]
  _Float16* hsh    = (_Float16*)(smem + B_ * GN_ * sizeof(_Float16)); // [32][512]

  const int dir  = blockIdx.x;
  const float*    gx  = gx2  + (size_t)dir * B_ * L_ * GN_;
  const _Float16* whh = whh2 + (size_t)dir * GN_ * H_;

  const int lane   = threadIdx.x & 31;
  const int wave   = threadIdx.x >> 5;
  const int mtile  = wave >> 4;          // 0..1  (batch tile)
  const int ntbase = (wave & 15) * 8;    // 8 N-tiles per wave

  // zero initial hidden state in LDS
  for (int i = threadIdx.x; i < B_ * H_; i += 1024) hsh[i] = (_Float16)0.f;

  // per-thread cell/hidden state registers: 16 contiguous (b,j) elements
  const int p0 = threadIdx.x * 16;
  const int bb = p0 >> 9;        // batch index
  const int j0 = p0 & 511;       // hidden index base
  float creg[16], hreg[16];
#pragma unroll
  for (int e = 0; e < 16; ++e) { creg[e] = 0.f; hreg[e] = 0.f; }

  __syncthreads();

  for (int t = 0; t < L_; ++t) {
    const int tt = (dir == 1) ? (L_ - 1 - t) : t;

    // ---- gates = gx[:,tt,:] + h @ Whh^T ----
    for (int i = 0; i < 8; ++i) {
      const int n0 = (ntbase + i) * 16;
      const _Float16* arow = hsh + (size_t)(mtile * 16 + (lane & 15)) * H_ + ((lane >> 4) * 16);
      const _Float16* wrow = whh + (size_t)(n0 + (lane & 15)) * H_ + ((lane >> 4) * 16);
      if (i < 7) __builtin_prefetch(wrow + 16 * H_, 0, 0);   // global_prefetch next N-tile
      v8f acc = {};
#pragma unroll
      for (int k0 = 0; k0 < H_; k0 += 32) {
        v16h a = *(const v16h*)(arow + k0);
        v16h b = *(const v16h*)(wrow + k0);
        acc = wmma_f16(a, b, acc);
      }
      const int col = n0 + (lane & 15);
#pragma unroll
      for (int v = 0; v < 8; ++v) {
        const int mrow = mtile * 16 + v + 8 * (lane >> 4);   // batch index
        float g = acc[v] + gx[((size_t)mrow * L_ + tt) * GN_ + col];
        gatesL[mrow * GN_ + col] = (_Float16)g;
      }
    }
    __syncthreads();

    // ---- pointwise LSTM cell ----
    const float mval = mask[bb * L_ + tt];
#pragma unroll
    for (int e = 0; e < 16; ++e) {
      const int j = j0 + e;
      const float gi = (float)gatesL[bb * GN_ + j];
      const float gf = (float)gatesL[bb * GN_ + H_ + j];
      const float gg = (float)gatesL[bb * GN_ + 2 * H_ + j];
      const float go = (float)gatesL[bb * GN_ + 3 * H_ + j];
      const float si = 1.f / (1.f + __expf(-gi));
      const float sf = 1.f / (1.f + __expf(-gf));
      const float so = 1.f / (1.f + __expf(-go));
      const float cn = sf * creg[e] + si * tanhf(gg);
      const float hn = so * tanhf(cn);
      hreg[e] += (hn - hreg[e]) * mval;
      creg[e] += (cn - creg[e]) * mval;
      hsh[bb * H_ + j] = (_Float16)hreg[e];
      outp[((size_t)bb * L_ + tt) * D2_ + dir * H_ + j] = (_Float16)hreg[e];
    }
    __syncthreads();
  }
}

// ---------------- fused vocab GEMM + log-softmax + argmax + weighted NLL ----------------
// One block per 16 rows; 8 waves each own disjoint vocab columns with online softmax stats.
__global__ __launch_bounds__(256) void vocab_loss_kernel(
    const _Float16* __restrict__ hid,   // [B*L][128]
    const _Float16* __restrict__ w2,    // [V][128]
    const float* __restrict__ b2, const int* __restrict__ tgt,
    const float* __restrict__ cew, float* __restrict__ outp, float* __restrict__ accum)
{
  __shared__ _Float16 hidL[16 * 128];
  __shared__ float patch[8 * 16 * 16];
  __shared__ float stats[8 * 16 * 5];

  const int tid  = threadIdx.x;
  const int lane = tid & 31;
  const int wave = tid >> 5;
  const int m0   = blockIdx.x * 16;

  for (int i = tid; i < 16 * 128; i += 256) hidL[i] = hid[(size_t)m0 * 128 + i];
  __syncthreads();

  float rm = -1e30f, rs = 0.f, rb = -1e30f, rt = -1e30f;
  int rbi = 0;
  const int myTgt = (lane < 16) ? tgt[m0 + lane] : -1;

  const int iters = V_ / (16 * 8);   // 250
  for (int it = 0; it < iters; ++it) {
    const int n0 = (it * 8 + wave) * 16;
    const _Float16* arow = hidL + (lane & 15) * 128 + ((lane >> 4) * 16);
    const _Float16* wrow = w2 + (size_t)(n0 + (lane & 15)) * 128 + ((lane >> 4) * 16);
    v8f acc = {};
#pragma unroll
    for (int k0 = 0; k0 < 128; k0 += 32) {
      v16h a = *(const v16h*)(arow + k0);
      v16h b = *(const v16h*)(wrow + k0);
      acc = wmma_f16(a, b, acc);
    }
    const float bbias = b2[n0 + (lane & 15)];
#pragma unroll
    for (int v = 0; v < 8; ++v) {
      const int r = v + 8 * (lane >> 4);
      patch[wave * 256 + r * 16 + (lane & 15)] = acc[v] + bbias;
    }
    __syncthreads();
    if (lane < 16) {
      for (int c = 0; c < 16; ++c) {
        const float x = patch[wave * 256 + lane * 16 + c];
        const int idx = n0 + c;
        if (x > rb) { rb = x; rbi = idx; }
        if (idx == myTgt) rt = x;
        if (x > rm) { rs = rs * __expf(rm - x) + 1.f; rm = x; }
        else        { rs += __expf(x - rm); }
      }
    }
    __syncthreads();
  }

  if (lane < 16) {
    float* st = stats + (wave * 16 + lane) * 5;
    st[0] = rm; st[1] = rs; st[2] = rb; st[3] = (float)rbi; st[4] = rt;
  }
  __syncthreads();

  if (tid < 16) {
    float M = -1e30f, S = 0.f, Bst = -1e30f, T = -1e30f;
    int BI = 0;
    for (int w = 0; w < 8; ++w) {
      const float* st = stats + (w * 16 + tid) * 5;
      const float mw = st[0], sw = st[1], bw = st[2], tw = st[4];
      const int biw = (int)st[3];
      const float nm = M > mw ? M : mw;
      S = S * __expf(M - nm) + sw * __expf(mw - nm);
      M = nm;
      if (bw > Bst) { Bst = bw; BI = biw; }
      if (tw > T) T = tw;
    }
    const float lse = M + __logf(S);
    outp[1 + m0 + tid] = (float)BI;                 // rcv_word
    const int t = tgt[m0 + tid];
    const float w = cew[t];
    atomicAdd(&accum[0], (lse - T) * w);            // sum nll*w
    atomicAdd(&accum[1], w);                        // sum w
  }
}

// ---------------- host-side launch sequence ----------------

extern "C" void kernel_launch(void* const* d_in, const int* in_sizes, int n_in,
                              void* d_out, int out_size, void* d_ws, size_t ws_size,
                              hipStream_t stream) {
  const int*   inp_word = (const int*)d_in[0];
  const float* inp_mask = (const float*)d_in[1];
  const int*   tgt_word = (const int*)d_in[3];
  const float* emb      = (const float*)d_in[4];
  const float* w_ih0    = (const float*)d_in[5];
  const float* w_hh0    = (const float*)d_in[6];
  const float* b0       = (const float*)d_in[7];
  const float* w_ih     = (const float*)d_in[8];
  const float* w_hh     = (const float*)d_in[9];
  const float* bvec     = (const float*)d_in[10];
  const float* dec_w1   = (const float*)d_in[11];
  const float* dec_b1   = (const float*)d_in[12];
  const float* dec_w2   = (const float*)d_in[13];
  const float* dec_b2   = (const float*)d_in[14];
  const float* ce_w     = (const float*)d_in[15];
  float* out = (float*)d_out;

  char* p = (char*)d_ws;
  auto alloc = [&](size_t bytes) -> char* {
    char* r = p; p += (bytes + 255) & ~(size_t)255; return r;
  };
  float*    accum = (float*)   alloc(2 * sizeof(float));
  _Float16* xemb  = (_Float16*)alloc((size_t)B_ * L_ * E_  * 2);
  _Float16* seqX  = (_Float16*)alloc((size_t)B_ * L_ * D2_ * 2);
  _Float16* seqY  = (_Float16*)alloc((size_t)B_ * L_ * D2_ * 2);
  float*    gx    = (float*)   alloc((size_t)2 * B_ * L_ * GN_ * 4);
  _Float16* wih0h = (_Float16*)alloc((size_t)2 * GN_ * E_  * 2);
  _Float16* whh0h = (_Float16*)alloc((size_t)2 * GN_ * H_  * 2);
  _Float16* wihh  = (_Float16*)alloc((size_t)4 * GN_ * D2_ * 2);
  _Float16* whhh  = (_Float16*)alloc((size_t)4 * GN_ * H_  * 2);
  _Float16* w1h   = (_Float16*)alloc((size_t)128 * D2_ * 2);
  _Float16* w2h   = (_Float16*)alloc((size_t)V_ * E_ * 2);
  _Float16* hidh  = (_Float16*)alloc((size_t)B_ * L_ * 128 * 2);

  init_accum_kernel<<<1, 1, 0, stream>>>(accum);

  auto conv = [&](const float* src, _Float16* dst, size_t n) {
    f32_to_f16_kernel<<<(int)((n + 255) / 256), 256, 0, stream>>>(src, dst, (int)n);
  };
  conv(w_ih0,  wih0h, (size_t)2 * GN_ * E_);
  conv(w_hh0,  whh0h, (size_t)2 * GN_ * H_);
  conv(w_ih,   wihh,  (size_t)4 * GN_ * D2_);
  conv(w_hh,   whhh,  (size_t)4 * GN_ * H_);
  conv(dec_w1, w1h,   (size_t)128 * D2_);
  conv(dec_w2, w2h,   (size_t)V_ * E_);

  embed_f16_kernel<<<(B_ * L_ * E_ + 255) / 256, 256, 0, stream>>>(inp_word, emb, xemb, B_ * L_, E_);

  const size_t recLds = (size_t)B_ * GN_ * 2 + (size_t)B_ * H_ * 2;  // 160 KB

  // ---- layer 0 ----
  for (int dir = 0; dir < 2; ++dir)
    gemm_f16_kernel<<<dim3(B_ * L_ / 16, GN_ / 128), 256, 0, stream>>>(
        xemb, wih0h + (size_t)dir * GN_ * E_, b0 + (size_t)dir * GN_,
        gx + (size_t)dir * B_ * L_ * GN_, nullptr, B_ * L_, GN_, E_, 0);
  lstm_rec_kernel<<<2, 1024, recLds, stream>>>(gx, whh0h, inp_mask, seqX);

  // ---- layer 1 ----
  for (int dir = 0; dir < 2; ++dir)
    gemm_f16_kernel<<<dim3(B_ * L_ / 16, GN_ / 128), 256, 0, stream>>>(
        seqX, wihh + (size_t)dir * GN_ * D2_, bvec + (size_t)dir * GN_,
        gx + (size_t)dir * B_ * L_ * GN_, nullptr, B_ * L_, GN_, D2_, 0);
  lstm_rec_kernel<<<2, 1024, recLds, stream>>>(gx, whhh, inp_mask, seqY);

  // ---- layer 2 ----
  for (int dir = 0; dir < 2; ++dir)
    gemm_f16_kernel<<<dim3(B_ * L_ / 16, GN_ / 128), 256, 0, stream>>>(
        seqY, wihh + (size_t)(2 + dir) * GN_ * D2_, bvec + (size_t)(2 + dir) * GN_,
        gx + (size_t)dir * B_ * L_ * GN_, nullptr, B_ * L_, GN_, D2_, 0);
  lstm_rec_kernel<<<2, 1024, recLds, stream>>>(gx, whhh + (size_t)2 * GN_ * H_, inp_mask, seqX);

  // ---- decoder hidden: relu(seq @ dec_w1^T + b1) ----
  gemm_f16_kernel<<<dim3(B_ * L_ / 16, 1), 256, 0, stream>>>(
      seqX, w1h, dec_b1, nullptr, hidh, B_ * L_, 128, D2_, 1);

  // ---- fused vocab GEMM + loss + argmax ----
  vocab_loss_kernel<<<B_ * L_ / 16, 256, 0, stream>>>(hidh, w2h, dec_b2, tgt_word, ce_w, out, accum);

  finalize_kernel<<<1, 1, 0, stream>>>(out, accum);
}
